// SimpleLSTM_3788161155239
// MI455X (gfx1250) — compile-verified, matrix operand order checked by reading, not run on hardware
//
#include <hip/hip_runtime.h>

#define H_ 64
#define T_ 512
#define B_ 2048
#define BT 16            // batch tile per workgroup
#define NWAVE 4
#define THREADS (NWAVE * 32)

typedef __attribute__((ext_vector_type(16))) __bf16 v16bf;
typedef __attribute__((ext_vector_type(8)))  float  v8f;

static __device__ __forceinline__ __bf16 f2bf(float f) {
  union { float f; unsigned u; } a; a.f = f;
  unsigned r = a.u + 0x7FFFu + ((a.u >> 16) & 1u);   // round-to-nearest-even
  union { unsigned short s; __bf16 b; } o; o.s = (unsigned short)(r >> 16);
  return o.b;
}

static __device__ __forceinline__ float sigm(float x) {
  return 1.0f / (1.0f + __expf(-x));
}
static __device__ __forceinline__ float tanh_f(float x) {
  x = fminf(fmaxf(x, -12.0f), 12.0f);
  float e = __expf(2.0f * x);
  return (e - 1.0f) / (e + 1.0f);
}

// K index held by (lane, element e) in the 16-bit A/B WMMA operand layout
// (A 16x32: lanes 0-15 -> K {0..7,16..23}, lanes 16-31 -> K {8..15,24..31};
//  B 32x16 mirrors with lane = N).
static __device__ __forceinline__ int k_of(int lane, int e) {
  int v = e >> 1, h = e & 1;
  int b = (lane >= 16) ? 8 : 0;
  return (v < 4) ? (2 * v + h + b) : (2 * (v - 4) + h + 16 + b);
}

static __device__ __forceinline__ v8f wmma_bf16(v16bf a, v16bf b, v8f c) {
  return __builtin_amdgcn_wmma_f32_16x16x32_bf16(false, a, false, b, (short)0, c,
                                                 false, false);
}

__global__ __launch_bounds__(THREADS)
void lstm2_fused_wmma(const float* __restrict__ x,
                      const float* __restrict__ Wih0, const float* __restrict__ Whh0,
                      const float* __restrict__ bih0, const float* __restrict__ bhh0,
                      const float* __restrict__ Wih1, const float* __restrict__ Whh1,
                      const float* __restrict__ bih1, const float* __restrict__ bhh1,
                      const float* __restrict__ Wfc,  const float* __restrict__ bfc,
                      float* __restrict__ out)
{
  // Packed bf16 B-matrix tiles: [weight][gate][wave][ktile][lane][elem]  (96 KB)
  __shared__ __align__(32) __bf16 wb[3][4][NWAVE][2][32][16];
  // h state in A-matrix layout: [layer][ktile][lane][elem]               (4 KB)
  __shared__ __align__(32) __bf16 hb[2][2][32][16];
  __shared__ __align__(16) float xs[BT * T_];                          // (32 KB)
  __shared__ float partial[NWAVE][BT];

  const int tid   = threadIdx.x;
  const int w     = tid >> 5;        // wave id: owns hidden units [16w, 16w+16)
  const int lane  = tid & 31;
  const int nl    = lane & 15;       // N (gate column) within tile
  const int halfu = lane >> 4;
  const int b0    = blockIdx.x * BT;

  // ---- one-time: pack Whh0 / Wih1 / Whh1 (transposed) into WMMA-B layout ----
  {
    const float* Ws[3] = { Whh0, Wih1, Whh1 };
    #pragma unroll
    for (int wt = 0; wt < 3; ++wt) {
      const float* W = Ws[wt];
      #pragma unroll
      for (int g = 0; g < 4; ++g)
        #pragma unroll
        for (int kk = 0; kk < 2; ++kk) {
          const int n = 64 * g + 16 * w + nl;     // gate row of W (= N of B)
          #pragma unroll
          for (int e = 0; e < 16; ++e) {
            const int k = 32 * kk + k_of(lane, e);
            wb[wt][g][w][kk][lane][e] = f2bf(W[n * H_ + k]);
          }
        }
    }
  }

  // zero h/c state buffers (h0 == h1 == 0 at t = 0)
  {
    unsigned* hz = (unsigned*)&hb[0][0][0][0];
    for (int i = tid; i < (int)(sizeof(hb) / 4); i += THREADS) hz[i] = 0u;
  }

  // preload this tile's x[b0:b0+16, :] (contiguous in [B,T,1]) into LDS
  {
    const float4* src = (const float4*)(x + (size_t)b0 * T_);
    float4* dst = (float4*)xs;
    for (int i = tid; i < BT * T_ / 4; i += THREADS) dst[i] = src[i];
  }

  // per-lane gate scalars (Wih0 is [256,1]; biases folded: bih + bhh)
  float wih0_l[4], bs0_l[4], bs1_l[4];
  #pragma unroll
  for (int g = 0; g < 4; ++g) {
    const int idx = 64 * g + 16 * w + nl;
    wih0_l[g] = Wih0[idx];
    bs0_l[g]  = bih0[idx] + bhh0[idx];
    bs1_l[g]  = bih1[idx] + bhh1[idx];
  }
  const float wfc_l = Wfc[16 * w + nl];

  __syncthreads();

  // hoist all 24 B tiles into registers: zero LDS/global weight traffic in loop
  v16bf Bhh0[4][2], Bih1[4][2], Bhh1[4][2];
  #pragma unroll
  for (int g = 0; g < 4; ++g)
    #pragma unroll
    for (int kk = 0; kk < 2; ++kk) {
      Bhh0[g][kk] = *(const v16bf*)&wb[0][g][w][kk][lane][0];
      Bih1[g][kk] = *(const v16bf*)&wb[1][g][w][kk][lane][0];
      Bhh1[g][kk] = *(const v16bf*)&wb[2][g][w][kk][lane][0];
    }

  // constants for scattering D-layout h back into A-layout LDS
  const int kglob = 16 * w + nl;           // global K index of this lane's unit
  const int kk_s  = kglob >> 5;
  const int kin   = kglob & 31;
  const int hi_s  = (kin >> 3) & 1;
  const int kr    = kin - (hi_s << 3);
  const int v_s   = (kr < 8) ? (kr >> 1) : (4 + ((kr - 16) >> 1));
  const int e_s   = (v_s << 1) | (kin & 1);

  v8f c0v = {0, 0, 0, 0, 0, 0, 0, 0};
  v8f c1v = {0, 0, 0, 0, 0, 0, 0, 0};
  float h1last[8];
  #pragma unroll
  for (int r = 0; r < 8; ++r) h1last[r] = 0.0f;

  for (int t = 0; t < T_; ++t) {
    // A operands: h0_{t-1}, h1_{t-1} (must be read before anyone overwrites)
    v16bf a00 = *(const v16bf*)&hb[0][0][lane][0];
    v16bf a01 = *(const v16bf*)&hb[0][1][lane][0];
    v16bf a10 = *(const v16bf*)&hb[1][0][lane][0];
    v16bf a11 = *(const v16bf*)&hb[1][1][lane][0];

    float xv[8];
    #pragma unroll
    for (int r = 0; r < 8; ++r) xv[r] = xs[(r + 8 * halfu) * T_ + t];

    // ---- layer 0: gates = x*Wih0 + b + h0 @ Whh0^T ----
    v8f acc[4];
    #pragma unroll
    for (int g = 0; g < 4; ++g) {
      v8f c;
      #pragma unroll
      for (int r = 0; r < 8; ++r) c[r] = fmaf(xv[r], wih0_l[g], bs0_l[g]);
      c = wmma_bf16(a00, Bhh0[g][0], c);
      c = wmma_bf16(a01, Bhh0[g][1], c);
      acc[g] = c;
    }

    float h0v[8];
    #pragma unroll
    for (int r = 0; r < 8; ++r) {
      const float iv = sigm(acc[0][r]);
      const float fv = sigm(acc[1][r]);
      const float gv = tanh_f(acc[2][r]);
      const float ov = sigm(acc[3][r]);
      const float cc = fv * c0v[r] + iv * gv;
      c0v[r] = cc;
      h0v[r] = ov * tanh_f(cc);
    }

    __syncthreads();                       // all waves done reading hb
    #pragma unroll
    for (int r = 0; r < 8; ++r) {          // h0_t -> A-layout LDS (bf16)
      const int laneA = (r + 8 * halfu) + (hi_s << 4);
      hb[0][kk_s][laneA][e_s] = f2bf(h0v[r]);
    }
    __syncthreads();                       // h0_t visible to all waves

    v16bf ax0 = *(const v16bf*)&hb[0][0][lane][0];
    v16bf ax1 = *(const v16bf*)&hb[0][1][lane][0];

    // ---- layer 1: gates = b + h0_t @ Wih1^T + h1_{t-1} @ Whh1^T ----
    #pragma unroll
    for (int g = 0; g < 4; ++g) {
      v8f c;
      #pragma unroll
      for (int r = 0; r < 8; ++r) c[r] = bs1_l[g];
      c = wmma_bf16(ax0, Bih1[g][0], c);
      c = wmma_bf16(ax1, Bih1[g][1], c);
      c = wmma_bf16(a10, Bhh1[g][0], c);
      c = wmma_bf16(a11, Bhh1[g][1], c);
      acc[g] = c;
    }

    float h1v[8];
    #pragma unroll
    for (int r = 0; r < 8; ++r) {
      const float iv = sigm(acc[0][r]);
      const float fv = sigm(acc[1][r]);
      const float gv = tanh_f(acc[2][r]);
      const float ov = sigm(acc[3][r]);
      const float cc = fv * c1v[r] + iv * gv;
      c1v[r] = cc;
      const float hh = ov * tanh_f(cc);
      h1v[r]    = hh;
      h1last[r] = hh;                      // fp32 copy of last h1 for the FC
    }

    #pragma unroll
    for (int r = 0; r < 8; ++r) {
      const int laneA = (r + 8 * halfu) + (hi_s << 4);
      hb[1][kk_s][laneA][e_s] = f2bf(h1v[r]);
    }
    __syncthreads();                       // h1_t committed before next read
  }

  // ---- FC on last timestep: out[m] = sum_j h1[m,j] * Wfc[j] + bfc ----
  float ps[8];
  #pragma unroll
  for (int r = 0; r < 8; ++r) ps[r] = h1last[r] * wfc_l;
  #pragma unroll
  for (int m = 1; m < 16; m <<= 1)
    #pragma unroll
    for (int r = 0; r < 8; ++r) ps[r] += __shfl_xor(ps[r], m, 16);
  if (nl == 0) {
    #pragma unroll
    for (int r = 0; r < 8; ++r) partial[w][r + 8 * halfu] = ps[r];
  }
  __syncthreads();
  if (tid < BT) {
    float s = bfc[0];
    #pragma unroll
    for (int ww = 0; ww < NWAVE; ++ww) s += partial[ww][tid];
    out[b0 + tid] = s;
  }
}

extern "C" void kernel_launch(void* const* d_in, const int* in_sizes, int n_in,
                              void* d_out, int out_size, void* d_ws, size_t ws_size,
                              hipStream_t stream) {
  (void)in_sizes; (void)n_in; (void)d_ws; (void)ws_size; (void)out_size;
  const float* x    = (const float*)d_in[0];
  const float* Wih0 = (const float*)d_in[1];
  const float* Whh0 = (const float*)d_in[2];
  const float* bih0 = (const float*)d_in[3];
  const float* bhh0 = (const float*)d_in[4];
  const float* Wih1 = (const float*)d_in[5];
  const float* Whh1 = (const float*)d_in[6];
  const float* bih1 = (const float*)d_in[7];
  const float* bhh1 = (const float*)d_in[8];
  const float* Wfc  = (const float*)d_in[9];
  const float* bfc  = (const float*)d_in[10];
  float* out = (float*)d_out;

  dim3 grid(B_ / BT), block(THREADS);
  hipLaunchKernelGGL(lstm2_fused_wmma, grid, block, 0, stream,
                     x, Wih0, Whh0, bih0, bhh0, Wih1, Whh1, bih1, bhh1,
                     Wfc, bfc, out);
}
